// EngineeringGNN_27444841021688
// MI455X (gfx1250) — matrix operand order repaired and checked by reading, not compiled, not run on hardware
//
#include <hip/hip_runtime.h>
#include <hip/hip_bf16.h>

typedef __attribute__((ext_vector_type(16))) _Float16 v16h;
typedef __attribute__((ext_vector_type(2)))  _Float16 v2h;
typedef __attribute__((ext_vector_type(8)))  float    v8f;
typedef __attribute__((ext_vector_type(4)))  int      v4i;

#define NN   100000
#define EDGN 1600000
#define BB   16

// Optional CDNA5 async global->LDS staging path
#if defined(__has_builtin)
#  if __has_builtin(__builtin_amdgcn_global_load_async_to_lds_b128) && \
      __has_builtin(__builtin_amdgcn_s_wait_asynccnt)
#    define USE_ASYNC_LDS 1
#  endif
#endif

// ---------------- packed WMMA fragment layout helpers -------------------
// Fragments are stored so each lane reads 16 contiguous halves (32B):
//   slot = ((kstep*ntiles + ctile)*32 + lane)*16 + i
// B-layout (ISA): col = ct*16 + (lane&15); K = kstep*32 + (lane&16?8:0) + (i&7) + (i&8?16:0)
// A-layout (ISA): row = lane&15;
//   lanes<16 : i0-7 -> K=kb+0..7,  i8-15 -> K=kb+16..23
//   lanes>=16: i0-7 -> K=kb+8..15, i8-15 -> K=kb+24..31   (kb = kstep*32)

__device__ __forceinline__ v16h frag_a_pk(const _Float16* sm, int kstep) {
  int lane = threadIdx.x & 31;
  return *(const v16h*)(sm + ((kstep * 32 + lane) << 4));
}
__device__ __forceinline__ v16h frag_b_pk(const _Float16* w, int ntiles, int kstep, int ct) {
  int lane = threadIdx.x & 31;
  return *(const v16h*)(w + (((kstep * ntiles + ct) * 32 + lane) << 4));
}
__device__ __forceinline__ v8f wmma32(v16h a, v16h b, v8f c) {
  return __builtin_amdgcn_wmma_f32_16x16x32_f16(false, a, false, b, (short)0, c,
                                                false, false);
}

// inverse A map: element (row r, K index c) -> (slot, i) within packed tile
__device__ __forceinline__ int a_slot(int r, int c) {
  int ks = c >> 5, kc = c & 31;
  return ks * 32 + (((kc >> 3) & 1) ? 16 : 0) + r;
}
__device__ __forceinline__ int a_elem(int c) {
  int kc = c & 31;
  return (kc & 7) + ((kc >> 4) & 1) * 8;
}

// stage a flat block of halves from global into LDS
__device__ __forceinline__ void stage_lds(const _Float16* __restrict__ g,
                                          _Float16* __restrict__ s, int halves) {
#ifdef USE_ASYNC_LDS
  for (int i = threadIdx.x * 8; i < halves; i += 256 * 8) {
    __builtin_amdgcn_global_load_async_to_lds_b128(
        (__attribute__((address_space(1))) v4i*)(g + i),
        (__attribute__((address_space(3))) v4i*)(s + i), 0, 0);
  }
#else
  const unsigned* gu = (const unsigned*)g;
  unsigned* su = (unsigned*)s;
  for (int i = threadIdx.x; i < (halves >> 1); i += 256) su[i] = gu[i];
#endif
}
__device__ __forceinline__ void stage_wait() {
#ifdef USE_ASYNC_LDS
  __builtin_amdgcn_s_wait_asynccnt(0);
#endif
}

// order-preserving float<->uint encode for atomic min/max
__device__ __forceinline__ unsigned fenc(float f) {
  unsigned u = __float_as_uint(f);
  return (u & 0x80000000u) ? ~u : (u | 0x80000000u);
}
__device__ __forceinline__ float fdec(unsigned e) {
  unsigned u = (e & 0x80000000u) ? (e ^ 0x80000000u) : ~e;
  return __uint_as_float(u);
}

// -------- pack f32 weights (K x ncols, zero-pad K beyond validK) --------
__global__ void k_pack_b(const float* __restrict__ src, _Float16* __restrict__ dst,
                         int validK, int ncols, int K) {
  int s = blockIdx.x * 256 + threadIdx.x;
  int tot = (K >> 5) * (ncols >> 4) * 512;
  if (s >= tot) return;
  int i = s & 15;
  int lane = (s >> 4) & 31;
  int tile = s >> 9;
  int ntiles = ncols >> 4;
  int ct = tile % ntiles, ks = tile / ntiles;
  int col = ct * 16 + (lane & 15);
  int kk = ks * 32 + ((lane & 16) ? 8 : 0) + (i & 7) + ((i & 8) ? 16 : 0);
  dst[s] = (kk < validK) ? (_Float16)src[(size_t)kk * ncols + col] : (_Float16)0.f;
}

__global__ void k_init(unsigned* __restrict__ zmin, unsigned* __restrict__ zmax) {
  int t = threadIdx.x;
  if (t < BB) { zmin[t] = 0xFFFFFFFFu; zmax[t] = 0u; }
}

// ---------- encoder: out = LN(relu(in@W1+b1)@W2+b2), 16 rows / block ----------
__global__ void k_encode(const float* __restrict__ in, int in_dim,
                         const _Float16* __restrict__ w1p,  // packed, K=32 x 128
                         const float* __restrict__ b1,
                         const _Float16* __restrict__ w2,   // packed, 128 x 128
                         const float* __restrict__ b2,
                         const float* __restrict__ g, const float* __restrict__ be,
                         float* __restrict__ out32, _Float16* __restrict__ out16) {
  __shared__ _Float16 sA[16 * 32];      // packed A, 1 kstep
  __shared__ _Float16 sW1[32 * 128];
  __shared__ _Float16 sW2[128 * 128];
  __shared__ _Float16 sT[16 * 128];     // packed A for GEMM2
  __shared__ float    sH[16 * 128];
  __shared__ float    sMu[16], sRs[16];

  size_t r0 = (size_t)blockIdx.x * 16;
  int tid = threadIdx.x;

  stage_lds(w1p, sW1, 32 * 128);
  stage_lds(w2, sW2, 128 * 128);

  // stage input tile directly in packed-A order (pairs of columns)
  for (int p = tid; p < 16 * 16; p += 256) {
    int r = p >> 4, c = (p & 15) * 2;   // c even, c<32
    float f0 = (c < in_dim) ? in[(r0 + r) * in_dim + c] : 0.f;
    float f1 = (c + 1 < in_dim) ? in[(r0 + r) * in_dim + c + 1] : 0.f;
    v2h hv; hv[0] = (_Float16)f0; hv[1] = (_Float16)f1;
    *(v2h*)(sA + a_slot(r, c) * 16 + a_elem(c)) = hv;
  }
  stage_wait();
  __syncthreads();

  int lane = tid & 31, wv = tid >> 5;
  int col = wv * 16 + (lane & 15);
  int mb  = (lane & 16) ? 8 : 0;

  // GEMM1: K padded to 32 -> single k-step
  v8f acc = {};
  acc = wmma32(frag_a_pk(sA, 0), frag_b_pk(sW1, 8, 0, wv), acc);
  {
    int slotBase = a_slot(0, col) * 16 + a_elem(col);
#pragma unroll
    for (int r = 0; r < 8; ++r)
      sT[slotBase + (mb + r) * 16] = (_Float16)fmaxf(acc[r] + b1[col], 0.f);
  }
  __syncthreads();

  // GEMM2: K=128 -> 4 k-steps
  v8f acc2 = {};
#pragma unroll
  for (int k = 0; k < 4; ++k)
    acc2 = wmma32(frag_a_pk(sT, k), frag_b_pk(sW2, 8, k, wv), acc2);
#pragma unroll
  for (int r = 0; r < 8; ++r)
    sH[(mb + r) * 128 + col] = acc2[r] + b2[col];
  __syncthreads();

  // LayerNorm
  if (tid < 16) {
    const float* row = sH + tid * 128;
    float m = 0.f;
    for (int c = 0; c < 128; ++c) m += row[c];
    m *= (1.f / 128.f);
    float v = 0.f;
    for (int c = 0; c < 128; ++c) { float d = row[c] - m; v += d * d; }
    v *= (1.f / 128.f);
    sMu[tid] = m;
    sRs[tid] = rsqrtf(v + 1e-5f);
  }
  __syncthreads();
  for (int i = tid; i < 16 * 128; i += 256) {
    int r = i >> 7, c = i & 127;
    float y = (sH[i] - sMu[r]) * sRs[r] * g[c] + be[c];
    size_t o = r0 * 128 + i;
    out16[o] = (_Float16)y;
    if (out32) out32[o] = y;
  }
}

// ---------- message + scatter: aggr[dst] += relu(h16[src] + e16) ----------
__global__ void k_msg(const _Float16* __restrict__ h16,
                      const _Float16* __restrict__ e16,
                      const int* __restrict__ ei, float* __restrict__ aggr) {
  int wv = threadIdx.x >> 5, lane = threadIdx.x & 31;
  size_t e = (size_t)blockIdx.x * 8 + wv;
  int s = ei[e], d = ei[(size_t)EDGN + e];
  size_t e2 = e + 8;
  if (e2 < (size_t)EDGN) __builtin_prefetch(h16 + (size_t)ei[e2] * 128, 0, 1);

  const _Float16* hp = h16 + (size_t)s * 128 + lane * 4;
  const _Float16* ep = e16 + e * 128 + lane * 4;
  float* ap = aggr + (size_t)d * 128 + lane * 4;
#pragma unroll
  for (int i = 0; i < 4; ++i) {
    float m = fmaxf((float)hp[i] + (float)ep[i], 0.f);
    __hip_atomic_fetch_add(ap + i, m, __ATOMIC_RELAXED, __HIP_MEMORY_SCOPE_AGENT);
  }
}

// ---------- conv layer: h = LN(h + relu(MLP(h + aggr))) ----------
__global__ void k_conv(float* __restrict__ h32, _Float16* __restrict__ h16,
                       const float* __restrict__ aggr,
                       const _Float16* __restrict__ w1, const float* __restrict__ b1,
                       const _Float16* __restrict__ w2, const float* __restrict__ b2,
                       const float* __restrict__ g, const float* __restrict__ be) {
  __shared__ _Float16 sA[16 * 128];     // packed A, 4 ksteps
  __shared__ float    sHin[16 * 128];
  __shared__ _Float16 sW1[128 * 128];
  __shared__ _Float16 sW2[128 * 128];
  __shared__ _Float16 sT[16 * 128];     // packed A for GEMM2
  __shared__ float    sH[16 * 128];
  __shared__ float    sMu[16], sRs[16];

  size_t r0 = (size_t)blockIdx.x * 16;
  int tid = threadIdx.x;

  stage_lds(w1, sW1, 128 * 128);
  stage_lds(w2, sW2, 128 * 128);

  for (int p = tid; p < 16 * 64; p += 256) {
    int r = p >> 6, c = (p & 63) * 2;   // pair of columns
    size_t o = r0 * 128 + r * 128 + c;
    float h0 = h32[o], h1 = h32[o + 1];
    sHin[r * 128 + c] = h0;
    sHin[r * 128 + c + 1] = h1;
    v2h hv; hv[0] = (_Float16)(h0 + aggr[o]); hv[1] = (_Float16)(h1 + aggr[o + 1]);
    *(v2h*)(sA + a_slot(r, c) * 16 + a_elem(c)) = hv;
  }
  stage_wait();
  __syncthreads();

  int lane = tid & 31, wv = tid >> 5;
  int col = wv * 16 + (lane & 15);
  int mb  = (lane & 16) ? 8 : 0;

  v8f acc = {};
#pragma unroll
  for (int k = 0; k < 4; ++k)
    acc = wmma32(frag_a_pk(sA, k), frag_b_pk(sW1, 8, k, wv), acc);
  {
    int slotBase = a_slot(0, col) * 16 + a_elem(col);
#pragma unroll
    for (int r = 0; r < 8; ++r)
      sT[slotBase + (mb + r) * 16] = (_Float16)fmaxf(acc[r] + b1[col], 0.f);
  }
  __syncthreads();

  v8f acc2 = {};
#pragma unroll
  for (int k = 0; k < 4; ++k)
    acc2 = wmma32(frag_a_pk(sT, k), frag_b_pk(sW2, 8, k, wv), acc2);
#pragma unroll
  for (int r = 0; r < 8; ++r) {
    int m = mb + r;
    sH[m * 128 + col] = sHin[m * 128 + col] + fmaxf(acc2[r] + b2[col], 0.f);
  }
  __syncthreads();

  if (tid < 16) {
    const float* row = sH + tid * 128;
    float m = 0.f;
    for (int c = 0; c < 128; ++c) m += row[c];
    m *= (1.f / 128.f);
    float v = 0.f;
    for (int c = 0; c < 128; ++c) { float d = row[c] - m; v += d * d; }
    v *= (1.f / 128.f);
    sMu[tid] = m;
    sRs[tid] = rsqrtf(v + 1e-5f);
  }
  __syncthreads();
  for (int i = tid; i < 16 * 128; i += 256) {
    int r = i >> 7, c = i & 127;
    float y = (sH[i] - sMu[r]) * sRs[r] * g[c] + be[c];
    size_t o = r0 * 128 + i;
    h32[o] = y;
    h16[o] = (_Float16)y;
  }
}

// ---------- per-batch z range via encoded atomics ----------
__global__ void k_zrange(const float* __restrict__ pos, const int* __restrict__ batch,
                         unsigned* __restrict__ zmin, unsigned* __restrict__ zmax) {
  int i = blockIdx.x * 256 + threadIdx.x;
  if (i >= NN) return;
  unsigned en = fenc(pos[i * 3 + 2]);
  int b = batch[i];
  atomicMin(zmin + b, en);
  atomicMax(zmax + b, en);
}

// ---------- per-batch displacement-scale head (tiny) ----------
__global__ void k_dsg(const float* __restrict__ force, const float* __restrict__ mat,
                      const unsigned* __restrict__ zmin, const unsigned* __restrict__ zmax,
                      const float* __restrict__ log_base,
                      const float* __restrict__ w1, const float* __restrict__ b1,
                      const float* __restrict__ w2, const float* __restrict__ b2,
                      float* __restrict__ out_dsg, float* __restrict__ out_mean) {
  __shared__ float sD[BB];
  int t = threadIdx.x;
  if (t < BB) {
    float fx = force[t * 3], fy = force[t * 3 + 1], fz = force[t * 3 + 2];
    float fm = sqrtf(fx * fx + fy * fy + fz * fz);
    float Em = fmaxf(mat[t * 2], 1.f);
    float nu = fminf(fmaxf(mat[t * 2 + 1], 0.f), 0.49f);
    float gl = fmaxf(fdec(zmax[t]) - fdec(zmin[t]), 1e-6f);
    float f0 = logf(fm + 1.f), f1 = logf(Em + 1e-12f), f2 = nu, f3 = logf(gl + 1e-6f);
    float val = b2[0];
    for (int j = 0; j < 64; ++j) {
      float tt = f0 * w1[j] + f1 * w1[64 + j] + f2 * w1[128 + j] + f3 * w1[192 + j] + b1[j];
      val += fmaxf(tt, 0.f) * w2[j];
    }
    val = fminf(fmaxf(val, -8.f), 8.f);
    float base = 1e-5f + log1pf(expf(log_base[0]));
    float d = fmaxf(base * expf(val), 1e-5f);
    out_dsg[t] = d;
    sD[t] = d;
  }
  __syncthreads();
  if (t == 0) {
    float s = 0.f;
    for (int b = 0; b < BB; ++b) s += sD[b];
    out_mean[0] = s * (1.f / BB);
  }
}

// ---------- decoder heads (both share one WMMA pass) ----------
__global__ void k_heads(const _Float16* __restrict__ h16,
                        const _Float16* __restrict__ dw1, const float* __restrict__ db1,
                        const float* __restrict__ dw2, const float* __restrict__ db2,
                        const _Float16* __restrict__ sw1, const float* __restrict__ sb1,
                        const float* __restrict__ sw2, const float* __restrict__ sb2,
                        const int* __restrict__ batch, const float* __restrict__ dsg,
                        float* __restrict__ u, float* __restrict__ raw,
                        float* __restrict__ sArr, float* __restrict__ lsArr,
                        float* __restrict__ safArr) {
  __shared__ _Float16 sA[16 * 128];   // packed A, 4 ksteps
  __shared__ _Float16 sWd[128 * 64];  // packed
  __shared__ _Float16 sWs[128 * 64];  // packed
  __shared__ float    sT[16 * 128];   // cols 0..63: disp head, 64..127: stress head

  size_t r0 = (size_t)blockIdx.x * 16;
  int tid = threadIdx.x;

  stage_lds(dw1, sWd, 128 * 64);
  stage_lds(sw1, sWs, 128 * 64);

  // stage h16 tile into packed-A order (uint pairs: columns c, c+1)
  {
    const unsigned* hu = (const unsigned*)(h16 + r0 * 128);
    for (int p = tid; p < 16 * 64; p += 256) {
      int r = p >> 6, c = (p & 63) * 2;
      unsigned val = hu[p];
      *(unsigned*)((char*)sA + a_slot(r, c) * 32 + a_elem(c) * 2) = val;
    }
  }
  stage_wait();
  __syncthreads();

  int lane = tid & 31, wv = tid >> 5;
  bool isS = wv >= 4;
  int ct = wv & 3;
  const _Float16* W = isS ? sWs : sWd;
  const float* bb = isS ? sb1 : db1;

  v8f acc = {};
#pragma unroll
  for (int k = 0; k < 4; ++k)
    acc = wmma32(frag_a_pk(sA, k), frag_b_pk(W, 4, k, ct), acc);

  int col = ct * 16 + (lane & 15);
  int mb  = (lane & 16) ? 8 : 0;
#pragma unroll
  for (int r = 0; r < 8; ++r)
    sT[(mb + r) * 128 + (isS ? 64 : 0) + col] = fmaxf(acc[r] + bb[col], 0.f);
  __syncthreads();

  if (tid < 48) {
    int nl = tid / 3, c = tid % 3;
    size_t n = r0 + nl;
    float a = db2[c];
    for (int j = 0; j < 64; ++j) a += sT[nl * 128 + j] * dw2[j * 3 + c];
    float d = dsg[batch[n]];
    raw[n * 3 + c] = a;
    u[n * 3 + c] = a * d;
  }
  if (tid >= 64 && tid < 80) {
    int nl = tid - 64;
    size_t n = r0 + nl;
    float a = sb2[0];
    for (int j = 0; j < 64; ++j) a += sT[nl * 128 + 64 + j] * sw2[j];
    float ls = fminf(fmaxf(a, 0.f), 30.f);
    float s = expf(ls);
    lsArr[n] = ls;
    sArr[n] = s;
    safArr[n] = 2.5e8f / (s + 1e-8f);
  }
}

// ------------------------------- launch ---------------------------------
extern "C" void kernel_launch(void* const* d_in, const int* in_sizes, int n_in,
                              void* d_out, int out_size, void* d_ws, size_t ws_size,
                              hipStream_t stream) {
  (void)in_sizes; (void)n_in; (void)out_size; (void)ws_size;

  const float* x          = (const float*)d_in[0];
  const float* edge_attr  = (const float*)d_in[1];
  const int*   edge_index = (const int*)d_in[2];
  const int*   batch      = (const int*)d_in[3];
  const float* force      = (const float*)d_in[4];
  const float* mat        = (const float*)d_in[5];
  const float* pos        = (const float*)d_in[6];
  const float* ne_w1 = (const float*)d_in[7];
  const float* ne_b1 = (const float*)d_in[8];
  const float* ne_w2 = (const float*)d_in[9];
  const float* ne_b2 = (const float*)d_in[10];
  const float* ne_g  = (const float*)d_in[11];
  const float* ne_be = (const float*)d_in[12];
  const float* ee_w1 = (const float*)d_in[13];
  const float* ee_b1 = (const float*)d_in[14];
  const float* ee_w2 = (const float*)d_in[15];
  const float* ee_b2 = (const float*)d_in[16];
  const float* ee_g  = (const float*)d_in[17];
  const float* ee_be = (const float*)d_in[18];
  const float* conv_w1 = (const float*)d_in[19];
  const float* conv_b1 = (const float*)d_in[20];
  const float* conv_w2 = (const float*)d_in[21];
  const float* conv_b2 = (const float*)d_in[22];
  const float* pn_g = (const float*)d_in[23];
  const float* pn_b = (const float*)d_in[24];
  const float* dh_w1 = (const float*)d_in[25];
  const float* dh_b1 = (const float*)d_in[26];
  const float* dh_w2 = (const float*)d_in[27];
  const float* dh_b2 = (const float*)d_in[28];
  const float* sm_w1 = (const float*)d_in[29];
  const float* sm_b1 = (const float*)d_in[30];
  const float* sm_w2 = (const float*)d_in[31];
  const float* sm_b2 = (const float*)d_in[32];
  const float* log_base = (const float*)d_in[33];
  const float* sh_w1 = (const float*)d_in[34];
  const float* sh_b1 = (const float*)d_in[35];
  const float* sh_w2 = (const float*)d_in[36];
  const float* sh_b2 = (const float*)d_in[37];

  // workspace carve-up
  char* ws = (char*)d_ws;
  size_t off = 0;
  auto take = [&](size_t bytes) {
    void* p = ws + off;
    off = (off + bytes + 255) & ~(size_t)255;
    return p;
  };
  float*     h32  = (float*)take((size_t)NN * 128 * 4);
  _Float16*  h16  = (_Float16*)take((size_t)NN * 128 * 2);
  float*     aggr = (float*)take((size_t)NN * 128 * 4);
  _Float16*  e16  = (_Float16*)take((size_t)EDGN * 128 * 2);
  _Float16*  wNe1 = (_Float16*)take(32 * 128 * 2);
  _Float16*  wNe2 = (_Float16*)take(128 * 128 * 2);
  _Float16*  wEe1 = (_Float16*)take(32 * 128 * 2);
  _Float16*  wEe2 = (_Float16*)take(128 * 128 * 2);
  _Float16*  wC1  = (_Float16*)take(3 * 128 * 128 * 2);
  _Float16*  wC2  = (_Float16*)take(3 * 128 * 128 * 2);
  _Float16*  wDh  = (_Float16*)take(128 * 64 * 2);
  _Float16*  wSh  = (_Float16*)take(128 * 64 * 2);
  unsigned*  zmin = (unsigned*)take(BB * 4);
  unsigned*  zmax = (unsigned*)take(BB * 4);

  // output layout: u | raw_u | s | log_s | mean | dsg | safety
  float* O = (float*)d_out;
  float* o_u    = O;
  float* o_raw  = O + (size_t)3 * NN;
  float* o_s    = O + (size_t)6 * NN;
  float* o_ls   = O + (size_t)7 * NN;
  float* o_mean = O + (size_t)8 * NN;
  float* o_dsg  = O + (size_t)8 * NN + 1;
  float* o_saf  = O + (size_t)8 * NN + 1 + BB;

  // weight packing (f32 -> f16, WMMA fragment order)
  auto pack = [&](const float* s, _Float16* d, int validK, int ncols, int K) {
    int tot = (K >> 5) * (ncols >> 4) * 512;
    k_pack_b<<<(tot + 255) / 256, 256, 0, stream>>>(s, d, validK, ncols, K);
  };
  pack(ne_w1, wNe1, 12, 128, 32);
  pack(ne_w2, wNe2, 128, 128, 128);
  pack(ee_w1, wEe1, 6, 128, 32);
  pack(ee_w2, wEe2, 128, 128, 128);
  for (int i = 0; i < 3; ++i) {
    pack(conv_w1 + (size_t)i * 128 * 128, wC1 + (size_t)i * 128 * 128, 128, 128, 128);
    pack(conv_w2 + (size_t)i * 128 * 128, wC2 + (size_t)i * 128 * 128, 128, 128, 128);
  }
  pack(dh_w1, wDh, 128, 64, 128);
  pack(sh_w1, wSh, 128, 64, 128);
  k_init<<<1, 32, 0, stream>>>(zmin, zmax);

  // encoders
  k_encode<<<NN / 16, 256, 0, stream>>>(x, 12, wNe1, ne_b1, wNe2, ne_b2, ne_g, ne_be,
                                        h32, h16);
  k_encode<<<EDGN / 16, 256, 0, stream>>>(edge_attr, 6, wEe1, ee_b1, wEe2, ee_b2,
                                          ee_g, ee_be, nullptr, e16);

  // message-passing layers
  for (int i = 0; i < 3; ++i) {
    (void)hipMemsetAsync(aggr, 0, (size_t)NN * 128 * 4, stream);
    k_msg<<<EDGN / 8, 256, 0, stream>>>(h16, e16, edge_index, aggr);
    k_conv<<<NN / 16, 256, 0, stream>>>(h32, h16, aggr,
                                        wC1 + (size_t)i * 128 * 128, conv_b1 + i * 128,
                                        wC2 + (size_t)i * 128 * 128, conv_b2 + i * 128,
                                        pn_g + i * 128, pn_b + i * 128);
  }

  // per-batch scale
  k_zrange<<<(NN + 255) / 256, 256, 0, stream>>>(pos, batch, zmin, zmax);
  k_dsg<<<1, 32, 0, stream>>>(force, mat, zmin, zmax, log_base,
                              sm_w1, sm_b1, sm_w2, sm_b2, o_dsg, o_mean);

  // decoder heads
  k_heads<<<NN / 16, 256, 0, stream>>>(h16, wDh, dh_b1, dh_w2, dh_b2,
                                       wSh, sh_b1, sh_w2, sh_b2,
                                       batch, o_dsg, o_u, o_raw, o_s, o_ls, o_saf);
}
